// graph_transformer_layer_27255862460929
// MI455X (gfx1250) — matrix operand order repaired and checked by reading, not compiled, not run on hardware
//
#include <hip/hip_runtime.h>
#include <hip/hip_bf16.h>
#include <math.h>

// ---------------------------------------------------------------------------
// graph_transformer_layer on MI455X (gfx1250), bf16 WMMA path.
//   qx = x@wq ; kx = x@wk ; vx = x@wv            (3 GEMMs, bf16 WMMA, f32 acc)
//   logits = qx@kx^T / 64 ; attn = softmax(rows) (streamed, WMMA, 2 passes)
//   out = attn * vx + raw_x                      (elementwise, fused in pass 2)
// Workspace: qb (bf16, 32MB) | kb (bf16, 32MB) | vx (f32, 64MB)  = 128MB.
//
//  * GEMM: register double-buffered staging (next K-tile loads overlap WMMA).
//  * Attention: async global->LDS (ASYNCcnt) with double-buffered LDS tiles,
//    guarded by __has_builtin with a synchronous fallback.
// ---------------------------------------------------------------------------

typedef __attribute__((ext_vector_type(16))) __bf16 v16bf;
typedef __attribute__((ext_vector_type(8)))  float  v8f;
typedef int v4i __attribute__((vector_size(16)));   // matches builtin param type

union Frag {            // 8 VGPRs: bf16 element 2v/2v+1 live in VGPR v lo/hi
    uint4 q[2];
    v16bf v;
};

__device__ __forceinline__ unsigned short f2bf(float f) {
    union { float f; unsigned u; } c; c.f = f;
    unsigned u = c.u;
    return (unsigned short)((u + 0x7FFFu + ((u >> 16) & 1u)) >> 16); // RNE
}

__device__ __forceinline__ v8f wmma_bf16(const Frag& a, const Frag& b, v8f c) {
    return __builtin_amdgcn_wmma_f32_16x16x32_bf16(
        /*neg_a=*/false, a.v, /*neg_b=*/false, b.v,
        /*c_mod=*/(short)0, c, /*reuse_a=*/false, /*reuse_b=*/false);
}

#if __has_builtin(__builtin_amdgcn_global_load_async_to_lds_b128) && \
    __has_builtin(__builtin_amdgcn_s_wait_asynccnt)
#define HAVE_ASYNC_LDS 1
#else
#define HAVE_ASYNC_LDS 0
#endif

// 16-byte global -> LDS copy; async DMA (no VGPR round trip) when available.
__device__ __forceinline__ void copy16_g2l(unsigned short* lds, const unsigned short* g) {
#if HAVE_ASYNC_LDS
    __builtin_amdgcn_global_load_async_to_lds_b128(
        (__attribute__((address_space(1))) v4i*)(void*)g,
        (__attribute__((address_space(3))) v4i*)(void*)lds,
        /*imm offset=*/0, /*cpol=*/0);
#else
    *(uint4*)lds = *(const uint4*)g;
#endif
}

__device__ __forceinline__ void g2l_drain() {
#if HAVE_ASYNC_LDS
    __builtin_amdgcn_s_wait_asynccnt(0);
#endif
}

#define TM 128
#define TN 128
#define TK 32

// C[M,N] = A[M,K] * B[K,N]; A,B row-major fp32 in global, converted to bf16
// while staging to LDS. B is stored TRANSPOSED in LDS (bt[n][k]) so the B
// fragment (col n, K=8h..8h+7 and 16+8h..16+8h+7 per lane) is two contiguous ds_load_b128.
// Staging is register double-buffered: the K+1 tile's global loads are issued
// before the WMMA block so they overlap compute.
template <bool STORE_BF16>
__global__ __launch_bounds__(256) void gemm_xw(
    const float* __restrict__ A, const float* __restrict__ B,
    void* __restrict__ Cout, int Nn)
{
    __shared__ __align__(16) unsigned short a_s[TM * TK];   // [m][k]
    __shared__ __align__(16) unsigned short bt_s[TN * TK];  // [n][k] (B^T)

    const int tid  = threadIdx.x;
    const int lane = tid & 31;
    const int wv   = tid >> 5;        // 8 waves
    const int wr   = wv >> 1;         // 0..3 : 32-row slice
    const int wc   = wv & 1;          // 0..1 : 64-col slice
    const int n    = lane & 15;
    const int h    = lane >> 4;

    const int rowBase = blockIdx.y * TM;
    const int colBase = blockIdx.x * TN;

    v8f acc[2][4];
    for (int mt = 0; mt < 2; ++mt)
        for (int nt = 0; nt < 4; ++nt) acc[mt][nt] = (v8f){};

    float4 va[4], vb[4];
    auto load_tiles = [&](int kb) {
        #pragma unroll
        for (int i = 0; i < 4; ++i) {           // A: 128x32, 1024 float4
            int f = tid + i * 256;
            int r = f >> 3, c4 = (f & 7) << 2;
            va[i] = *(const float4*)&A[(size_t)(rowBase + r) * Nn + kb + c4];
        }
        #pragma unroll
        for (int i = 0; i < 4; ++i) {           // B: 32x128, 1024 float4
            int f = tid + i * 256;
            int kr = f >> 5, c4 = (f & 31) << 2;
            vb[i] = *(const float4*)&B[(size_t)(kb + kr) * Nn + colBase + c4];
        }
    };
    auto store_tiles = [&]() {
        #pragma unroll
        for (int i = 0; i < 4; ++i) {
            int f = tid + i * 256;
            int r = f >> 3, c4 = (f & 7) << 2;
            uint2 p;
            p.x = (unsigned)f2bf(va[i].x) | ((unsigned)f2bf(va[i].y) << 16);
            p.y = (unsigned)f2bf(va[i].z) | ((unsigned)f2bf(va[i].w) << 16);
            *(uint2*)&a_s[r * TK + c4] = p;
        }
        #pragma unroll
        for (int i = 0; i < 4; ++i) {
            int f = tid + i * 256;
            int kr = f >> 5, c4 = (f & 31) << 2;
            bt_s[(c4 + 0) * TK + kr] = f2bf(vb[i].x);
            bt_s[(c4 + 1) * TK + kr] = f2bf(vb[i].y);
            bt_s[(c4 + 2) * TK + kr] = f2bf(vb[i].z);
            bt_s[(c4 + 3) * TK + kr] = f2bf(vb[i].w);
        }
    };

    load_tiles(0);
    for (int kb = 0; kb < Nn; kb += TK) {
        store_tiles();
        __syncthreads();
        if (kb + TK < Nn) load_tiles(kb + TK);  // overlap next loads with WMMA

        Frag af[2];
        #pragma unroll
        for (int mt = 0; mt < 2; ++mt) {
            int r = wr * 32 + mt * 16 + (lane & 15);
            af[mt].q[0] = *(const uint4*)&a_s[r * TK + 8 * h];       // K=8h..8h+7
            af[mt].q[1] = *(const uint4*)&a_s[r * TK + 16 + 8 * h];  // K=16+8h..
        }
        #pragma unroll
        for (int nt = 0; nt < 4; ++nt) {
            Frag bf;
            int c = wc * 64 + nt * 16 + n;
            bf.q[0] = *(const uint4*)&bt_s[c * TK + 8 * h];          // K=8h..8h+7
            bf.q[1] = *(const uint4*)&bt_s[c * TK + 16 + 8 * h];     // K=16+8h..
            #pragma unroll
            for (int mt = 0; mt < 2; ++mt)
                acc[mt][nt] = wmma_bf16(af[mt], bf, acc[mt][nt]);
        }
        __syncthreads();
    }

    // ---- write C: lane holds (m = v + 8h, n = lane&15) per 16x16 tile -----
    for (int mt = 0; mt < 2; ++mt)
        for (int nt = 0; nt < 4; ++nt)
            #pragma unroll
            for (int v = 0; v < 8; ++v) {
                int row = rowBase + wr * 32 + mt * 16 + v + 8 * h;
                int col = colBase + wc * 64 + nt * 16 + n;
                float val = acc[mt][nt][v];
                if (STORE_BF16)
                    ((unsigned short*)Cout)[(size_t)row * Nn + col] = f2bf(val);
                else
                    ((float*)Cout)[(size_t)row * Nn + col] = val;
            }
}

// Fused attention: each wave owns 16 rows (WG = 128 rows). Two passes over
// 128-wide column blocks; logits tiles recomputed via WMMA each pass so the
// 64MB logits matrix is never materialized. kb is row-major [j][k], which is
// exactly the transposed-B LDS layout needed for S = q @ k^T.
// Staging uses double-buffered async global->LDS DMA overlapped with WMMA.
__global__ __launch_bounds__(256) void attn_fused(
    const unsigned short* __restrict__ qb, const unsigned short* __restrict__ kbm,
    const float* __restrict__ vx, const float* __restrict__ raw,
    float* __restrict__ out, int Nn)
{
    __shared__ __align__(16) unsigned short q_s[2][128 * TK];  // [r][k]
    __shared__ __align__(16) unsigned short k_s[2][128 * TK];  // [j][k] == B^T

    const int tid  = threadIdx.x;
    const int lane = tid & 31;
    const int wv   = tid >> 5;
    const int n    = lane & 15;
    const int h    = lane >> 4;

    const int rowBlock = blockIdx.x * 128;
    const int myRow    = rowBlock + wv * 16;      // wave's 16 rows
    const float scale  = 1.0f / 64.0f;            // 1/sqrt(4096)

    // stage (q rows [rowBlock,+128), k rows [jb,+128)), cols kk..kk+32
    auto stage = [&](int buf, int jb, int kk) {
        #pragma unroll
        for (int i = 0; i < 2; ++i) {
            int f = tid + i * 256;                // 0..511 uint4 per buffer
            int r = f >> 2, c8 = (f & 3) << 3;
            copy16_g2l(&q_s[buf][r * TK + c8],
                       &qb[(size_t)(rowBlock + r) * Nn + kk + c8]);
            copy16_g2l(&k_s[buf][r * TK + c8],
                       &kbm[(size_t)(jb + r) * Nn + kk + c8]);
        }
    };

    float rmax[8], rsum[8];
    for (int v = 0; v < 8; ++v) { rmax[v] = -INFINITY; rsum[v] = 0.0f; }

    for (int phase = 0; phase < 2; ++phase) {
        for (int jb = 0; jb < Nn; jb += 128) {
            v8f acc[8];
            #pragma unroll
            for (int nt = 0; nt < 8; ++nt) acc[nt] = (v8f){};

            stage(0, jb, 0);                      // prologue fill of buffer 0
            g2l_drain();
            __syncthreads();

            for (int kk = 0; kk < Nn; kk += TK) {
                const int cur = (kk >> 5) & 1;
                const int nxt = cur ^ 1;
                if (kk + TK < Nn) stage(nxt, jb, kk + TK);  // DMA overlaps WMMA

                const unsigned short* qs = q_s[cur];
                const unsigned short* ks = k_s[cur];
                Frag af;
                int r = wv * 16 + (lane & 15);
                af.q[0] = *(const uint4*)&qs[r * TK + 8 * h];
                af.q[1] = *(const uint4*)&qs[r * TK + 16 + 8 * h];
                #pragma unroll
                for (int nt = 0; nt < 8; ++nt) {
                    Frag bf;
                    int c = nt * 16 + n;
                    bf.q[0] = *(const uint4*)&ks[c * TK + 8 * h];
                    bf.q[1] = *(const uint4*)&ks[c * TK + 16 + 8 * h];
                    acc[nt] = wmma_bf16(af, bf, acc[nt]);
                }
                g2l_drain();                      // next buffer landed
                __syncthreads();                  // all waves done reading cur
            }

            #pragma unroll
            for (int nt = 0; nt < 8; ++nt)
                #pragma unroll
                for (int v = 0; v < 8; ++v) acc[nt][v] *= scale;

            if (phase == 0) {
                // online (max, sum-exp) update for this 128-col block
                #pragma unroll
                for (int v = 0; v < 8; ++v) {
                    float m = acc[0][v];
                    #pragma unroll
                    for (int nt = 1; nt < 8; ++nt) m = fmaxf(m, acc[nt][v]);
                    for (int off = 1; off < 16; off <<= 1)
                        m = fmaxf(m, __shfl_xor(m, off, 32));
                    float nm = fmaxf(rmax[v], m);
                    float s = 0.0f;
                    #pragma unroll
                    for (int nt = 0; nt < 8; ++nt)
                        s += __expf(acc[nt][v] - nm);
                    for (int off = 1; off < 16; off <<= 1)
                        s += __shfl_xor(s, off, 32);
                    rsum[v] = rsum[v] * __expf(rmax[v] - nm) + s;
                    rmax[v] = nm;
                }
            } else {
                // finalize: out = softmax(logits) * vx + raw  (elementwise)
                #pragma unroll
                for (int v = 0; v < 8; ++v) {
                    int row = myRow + v + 8 * h;
                    float inv = 1.0f / rsum[v];
                    #pragma unroll
                    for (int nt = 0; nt < 8; ++nt) {
                        int col = jb + nt * 16 + n;
                        size_t idx = (size_t)row * Nn + col;
                        float p = __expf(acc[nt][v] - rmax[v]) * inv;
                        out[idx] = fmaf(p, vx[idx], raw[idx]);
                    }
                }
            }
        }
    }
}

extern "C" void kernel_launch(void* const* d_in, const int* in_sizes, int n_in,
                              void* d_out, int out_size, void* d_ws, size_t ws_size,
                              hipStream_t stream) {
    (void)in_sizes; (void)n_in; (void)out_size; (void)ws_size;
    const int Nn = 4096;
    const float* x   = (const float*)d_in[0];
    const float* raw = (const float*)d_in[1];
    const float* wq  = (const float*)d_in[2];
    const float* wk  = (const float*)d_in[3];
    const float* wvp = (const float*)d_in[4];
    float* out = (float*)d_out;

    // workspace layout: qb(bf16 32MB) | kb(bf16 32MB) | vx(f32 64MB)
    unsigned short* qb = (unsigned short*)d_ws;
    unsigned short* kb = qb + (size_t)Nn * Nn;
    float* vx = (float*)((char*)d_ws + 2u * (size_t)Nn * Nn * sizeof(unsigned short));

    dim3 g(Nn / TN, Nn / TM);
    gemm_xw<true ><<<g, 256, 0, stream>>>(x, wq,  (void*)qb, Nn);
    gemm_xw<true ><<<g, 256, 0, stream>>>(x, wk,  (void*)kb, Nn);
    gemm_xw<false><<<g, 256, 0, stream>>>(x, wvp, (void*)vx, Nn);
    attn_fused<<<Nn / 128, 256, 0, stream>>>(qb, kb, vx, raw, out, Nn);
}
